// SimCLR_57097295233313
// MI455X (gfx1250) — matrix operand order repaired and checked by reading, not compiled, not run on hardware
//
#include <hip/hip_runtime.h>

typedef __attribute__((ext_vector_type(16))) _Float16 v16h;
typedef __attribute__((ext_vector_type(8)))  _Float16 v8h;
typedef __attribute__((ext_vector_type(8)))  float    v8f;

#define NROWS   8192          // 2N
#define HALFN   4096          // N
#define DDIM    256           // D
#define TILES   (NROWS / 16)  // 512 row/col tiles
#define KSTEPS  (DDIM / 32)   // 8 wmma per 16x16 tile
#define WAVES   8             // 256-thread block = 8 wave32

// ---------------------------------------------------------------------------
// Kernel 1: row-normalize concat(z_i, z_j) into f16 workspace
// ---------------------------------------------------------------------------
__global__ __launch_bounds__(256) void nrm_kernel(const float* __restrict__ zi,
                                                  const float* __restrict__ zj,
                                                  _Float16* __restrict__ zn) {
    __shared__ float red[256];
    const int row = blockIdx.x;
    const int tid = threadIdx.x;
    const float* src = (row < HALFN) ? (zi + (size_t)row * DDIM)
                                     : (zj + (size_t)(row - HALFN) * DDIM);
    float x = src[tid];
    red[tid] = x * x;
    __syncthreads();
    for (int s = 128; s > 0; s >>= 1) {
        if (tid < s) red[tid] += red[tid + s];
        __syncthreads();
    }
    float norm = fmaxf(sqrtf(red[0]), 1e-8f);
    zn[(size_t)row * DDIM + tid] = (_Float16)(x / norm);
}

// ---------------------------------------------------------------------------
// Kernel 2: positives — pos[i] = zn[i] . zn[(i+HALFN) % NROWS]
// (pulled out of the GEMM kernel so its inner loop is branch-free)
// ---------------------------------------------------------------------------
__global__ __launch_bounds__(256) void pos_kernel(const _Float16* __restrict__ zn,
                                                  float* __restrict__ pos) {
    __shared__ float red[256];
    const int row  = blockIdx.x;
    const int mate = (row + HALFN) & (NROWS - 1);
    const int tid  = threadIdx.x;
    float a = (float)zn[(size_t)row  * DDIM + tid];
    float b = (float)zn[(size_t)mate * DDIM + tid];
    red[tid] = a * b;
    __syncthreads();
    for (int s = 128; s > 0; s >>= 1) {
        if (tid < s) red[tid] += red[tid + s];
        __syncthreads();
    }
    if (tid == 0) pos[row] = red[0];
}

// ---------------------------------------------------------------------------
// Kernel 3: fused sim = zn@zn^T with online masked sum(exp) per row.
// One block per 16-row tile; each wave32 sweeps a strided set of 16-col tiles
// with V_WMMA_F32_16X16X32_F16. Diagonal masking is branch-free (cndmask).
// ---------------------------------------------------------------------------
__global__ __launch_bounds__(256) void sim_kernel(const _Float16* __restrict__ zn,
                                                  float* __restrict__ rowsum) {
    __shared__ float rs[16];
    const int it   = blockIdx.x;        // row-tile index
    const int lane = threadIdx.x & 31;
    const int wave = threadIdx.x >> 5;
    const int n    = lane & 15;         // local column held by this lane
    const int hi   = lane >> 4;         // 0: rows m=0..7, 1: rows m=8..15

    if (threadIdx.x < 16) rs[threadIdx.x] = 0.0f;
    __syncthreads();

    // Preload register-resident A fragments for this 16-row tile.
    // A order used here (contiguous k, B matches): lanes 0-15 hold row M=lane, K=k0..k0+15;
    // lanes 16-31 hold the same rows, K=k0+16..k0+31.
    const _Float16* arow = zn + (size_t)(it * 16 + n) * DDIM;
    v16h afrag[KSTEPS];
#pragma unroll
    for (int k = 0; k < KSTEPS; ++k) {
        v8h c0 = *(const v8h*)(arow + k * 32 + hi * 16);
        v8h c1 = *(const v8h*)(arow + k * 32 + hi * 16 + 8);
#pragma unroll
        for (int t = 0; t < 8; ++t) { afrag[k][t] = c0[t]; afrag[k][8 + t] = c1[t]; }
    }

    v8f rowacc = {};   // per-lane partial sum(exp) for 8 rows (this lane's cols)

    for (int jt = wave; jt < TILES; jt += WAVES) {
        // B order used here (contiguous k, matches A): lane holds column N=lane%16, 16 contiguous K
        // (low 16 lanes: K=0..15, high 16 lanes: K=16..31 of the 32-chunk).
        const _Float16* bcol = zn + (size_t)(jt * 16 + n) * DDIM + hi * 16;
        v8f c = {};
#pragma unroll
        for (int k = 0; k < KSTEPS; ++k) {
            v16h b = *(const v16h*)(bcol + k * 32);
            c = __builtin_amdgcn_wmma_f32_16x16x32_f16(
                    false, afrag[k], false, b, (short)0, c, false, false);
        }
        // C layout: VGPR v <-> row m = v + 8*hi, lane%16 <-> col n.
        // Exclude the global diagonal (self-similarity) branch-free.
#pragma unroll
        for (int v = 0; v < 8; ++v) {
            const float e    = __expf(c[v]);
            const bool  diag = (jt == it) & (n == (v + 8 * hi));
            rowacc[v] += diag ? 0.0f : e;
        }
    }

    // Reduce the 16 lanes holding the same rows (masks 1,2,4,8 stay in-half).
#pragma unroll
    for (int v = 0; v < 8; ++v) {
        float s = rowacc[v];
        s += __shfl_xor(s, 1);
        s += __shfl_xor(s, 2);
        s += __shfl_xor(s, 4);
        s += __shfl_xor(s, 8);
        rowacc[v] = s;
    }
    if (n == 0) {  // lanes 0 and 16 carry full sums for rows 0-7 / 8-15
#pragma unroll
        for (int v = 0; v < 8; ++v) atomicAdd(&rs[8 * hi + v], rowacc[v]);
    }
    __syncthreads();
    if (threadIdx.x < 16) rowsum[it * 16 + threadIdx.x] = rs[threadIdx.x];
}

// ---------------------------------------------------------------------------
// Kernel 4: loss = mean_i ( log(denom_i) - pos_i )   (TEMPERATURE = 1)
// ---------------------------------------------------------------------------
__global__ __launch_bounds__(256) void loss_kernel(const float* __restrict__ rowsum,
                                                   const float* __restrict__ pos,
                                                   float* __restrict__ out) {
    __shared__ float red[256];
    const int tid = threadIdx.x;
    float s = 0.0f;
    for (int i = tid; i < NROWS; i += 256) s += logf(rowsum[i]) - pos[i];
    red[tid] = s;
    __syncthreads();
    for (int st = 128; st > 0; st >>= 1) {
        if (tid < st) red[tid] += red[tid + st];
        __syncthreads();
    }
    if (tid == 0) out[0] = red[0] / (float)NROWS;
}

// ---------------------------------------------------------------------------
extern "C" void kernel_launch(void* const* d_in, const int* in_sizes, int n_in,
                              void* d_out, int out_size, void* d_ws, size_t ws_size,
                              hipStream_t stream) {
    const float* zi = (const float*)d_in[0];
    const float* zj = (const float*)d_in[1];

    _Float16* zn     = (_Float16*)d_ws;                                  // 4 MB
    float*    rowsum = (float*)((char*)d_ws + (size_t)NROWS * DDIM * 2); // 32 KB
    float*    pos    = rowsum + NROWS;                                   // 32 KB

    nrm_kernel <<<NROWS, 256, 0, stream>>>(zi, zj, zn);
    pos_kernel <<<NROWS, 256, 0, stream>>>(zn, pos);
    sim_kernel <<<TILES, 256, 0, stream>>>(zn, rowsum);
    loss_kernel<<<1,     256, 0, stream>>>(rowsum, pos, (float*)d_out);
}
